// NVGPTAttention_5111011082471
// MI455X (gfx1250) — matrix operand order repaired and checked by reading, not compiled, not run on hardware
//
#include <hip/hip_runtime.h>

// ---------------- problem constants ----------------
#define BB   2
#define SS   2048
#define HH   4096
#define NHH  32
#define HDD  128
#define N3   (3*HH)           // 12288
#define MM   (BB*SS)          // 4096
#define HEADS (BB*NHH)        // 64
#define PER_BUF 16777216      // B*NH*S*HD elements per Q/K/V buffer

typedef __attribute__((ext_vector_type(16))) __bf16 v16bf;
typedef __attribute__((ext_vector_type(8)))  __bf16 v8bf;
typedef __attribute__((ext_vector_type(8)))  float  v8f;
typedef __attribute__((ext_vector_type(4)))  unsigned int u32x4;
typedef __attribute__((ext_vector_type(8)))  int i32x8;
typedef __attribute__((ext_vector_type(4)))  int i32x4;

#if __has_builtin(__builtin_amdgcn_tensor_load_to_lds) && \
    __has_builtin(__builtin_amdgcn_s_wait_tensorcnt)
#define HAVE_TDM 1
#else
#define HAVE_TDM 0
#endif

// ---------------- helpers ----------------
static __device__ __forceinline__ __bf16 f2bf(float f) {
  unsigned u = __builtin_bit_cast(unsigned, f);
  unsigned rr = u + 0x7FFFu + ((u >> 16) & 1u);
  unsigned short h = (unsigned short)(rr >> 16);
  return __builtin_bit_cast(__bf16, h);
}
static __device__ __forceinline__ float bf2f(__bf16 v) {
  unsigned short h = __builtin_bit_cast(unsigned short, v);
  unsigned u = ((unsigned)h) << 16;
  return __builtin_bit_cast(float, u);
}
static __device__ __forceinline__ v8f v8f_zero() {
  v8f x;
#pragma unroll
  for (int i = 0; i < 8; ++i) x[i] = 0.0f;
  return x;
}
static __device__ __forceinline__ v16bf cat8(v8bf lo, v8bf hi) {
  v16bf r;
#pragma unroll
  for (int i = 0; i < 8; ++i) { r[i] = lo[i]; r[i + 8] = hi[i]; }
  return r;
}
static __device__ __forceinline__ v8f wmma32(v16bf a, v16bf b, v8f c) {
  // D(16x16,f32) = A(16x32,bf16) x B(32x16,bf16) + C
  return __builtin_amdgcn_wmma_f32_16x16x32_bf16(false, a, false, b, (short)0, c,
                                                 false, false);
}

#if HAVE_TDM
// 2D TDM: copy tile (tile0 x tile1 bf16 elems) from global (row stride
// `stride0` elems) into contiguous LDS at lds byte-offset `lds_off`.
// D# layout per CDNA5 ISA; data_size=1 (2 bytes).
// This toolchain exposes the 6-arg builtin:
//   (u32x4 g0, i32x8 g1, i32x4 g2, i32x4 g3, i32x8 g4, i32 cpol)
static __device__ __forceinline__ void tdm_load_2d(
    unsigned lds_off, const void* gptr,
    unsigned tile0, unsigned tile1,
    unsigned td0, unsigned td1, unsigned stride0) {
  unsigned long long ga = (unsigned long long)gptr;
  u32x4 g0 = {};
  g0[0] = 1u;                                            // count=1 (user D#)
  g0[1] = lds_off;                                       // lds_addr (bytes)
  g0[2] = (unsigned)(ga & 0xFFFFFFFFu);                  // global_addr[31:0]
  g0[3] = (unsigned)((ga >> 32) & 0x1FFFFFFu) | (2u << 30); // [56:32] | type=2
  i32x8 g1 = {};
  g1[0] = (int)(1u << 16);                               // data_size=1 -> 2B elems
  g1[1] = (int)((td0 & 0xFFFFu) << 16);                  // tensor_dim0[15:0]
  g1[2] = (int)(((td0 >> 16) & 0xFFFFu) | ((td1 & 0xFFFFu) << 16));
  g1[3] = (int)(((td1 >> 16) & 0xFFFFu) | ((tile0 & 0xFFFFu) << 16));
  g1[4] = (int)(tile1 & 0xFFFFu);                        // tile_dim1, tile_dim2=0
  g1[5] = (int)stride0;                                  // dim0_stride[31:0]
  g1[6] = 0;                                             // stride hi / dim1_stride lo
  g1[7] = 0;
  i32x4 g2 = {};
  i32x4 g3 = {};
  i32x8 g4 = {};
  __builtin_amdgcn_tensor_load_to_lds(g0, g1, g2, g3, g4, 0);
}
#endif

// ---------------- f32 -> bf16 convert ----------------
__global__ void k_cvt(const float* __restrict__ in, __bf16* __restrict__ out, int n) {
  int i = blockIdx.x * blockDim.x + threadIdx.x;
  if (i < n) out[i] = f2bf(in[i]);
}

// ---------------- shared 64x64 WMMA GEMM mainloop ----------------
// Block tile 64(M) x 64(N), BK=32, 8 waves in 4x2 (wave tile 16x32).
// As: [64 m][32 k] row-major (TDM-loaded). Bst: [64 n][32 k] (transposed).
static __device__ __forceinline__ void gemm_mainloop(
    const __bf16* __restrict__ A, const __bf16* __restrict__ Bm,
    int K, int N, int m_blk, int n_blk,
    __bf16* As, __bf16* Bst, v8f& acc0, v8f& acc1) {
  const int tid  = threadIdx.x;
  const int lane = tid & 31, w = tid >> 5;
  const int wm = w >> 1, wn = w & 1;
  const int br = tid >> 3, bc = (tid & 7) << 3;   // B stage: 32 rows x 8 groups of 8
  const int r = lane & 15, hf = lane >> 4;
  const int arow = wm * 16 + r;
  acc0 = v8f_zero();
  acc1 = v8f_zero();
  for (int kk = 0; kk < K; kk += 32) {
#if HAVE_TDM
    if (w == 0) {
      tdm_load_2d((unsigned)(size_t)As, A + (size_t)m_blk * K + kk,
                  /*tile*/32, 64, /*tensor*/32, 64, /*stride*/(unsigned)K);
    }
#else
    {
      const int ar = tid >> 2, ac = (tid & 3) << 3; // A stage: 64 rows x 4 groups
      uint4 av = *(const uint4*)(A + (size_t)(m_blk + ar) * K + (kk + ac));
      *(uint4*)(As + ar * 32 + ac) = av;
    }
#endif
    uint4 bv = *(const uint4*)(Bm + (size_t)(kk + br) * N + (n_blk + bc));
    union { uint4 u; __bf16 h[8]; } bu; bu.u = bv;
#pragma unroll
    for (int i = 0; i < 8; ++i) Bst[(bc + i) * 32 + br] = bu.h[i];
    if (kk + 32 < K)  // prime L2 for next k-step (global_prefetch_b8)
      __builtin_prefetch(Bm + (size_t)(kk + 32 + br) * N + (n_blk + bc), 0, 1);
#if HAVE_TDM
    if (w == 0) __builtin_amdgcn_s_wait_tensorcnt(0);
#endif
    __syncthreads();
    // A fragment: lane half hf covers k = hf*16..hf*16+15
    v16bf af = cat8(*(const v8bf*)(As + arow * 32 + hf * 16),
                    *(const v8bf*)(As + arow * 32 + hf * 16 + 8));
    const int n0 = wn * 32 + r;     // B fragment: n = lane%16, k = hf*16 + j
    const int kb = hf * 16;
    v16bf b0 = cat8(*(const v8bf*)(Bst + n0 * 32 + kb),
                    *(const v8bf*)(Bst + n0 * 32 + kb + 8));
    v16bf b1 = cat8(*(const v8bf*)(Bst + (n0 + 16) * 32 + kb),
                    *(const v8bf*)(Bst + (n0 + 16) * 32 + kb + 8));
    acc0 = wmma32(af, b0, acc0);
    acc1 = wmma32(af, b1, acc1);
    __syncthreads();
  }
}

// ---------------- QKV projection GEMM ----------------
// qkv base holds Q|K|V, each [B,NH,S,HD] bf16 (head-major for attention).
static __device__ __forceinline__ void qkv_scatter(__bf16* qkv, int row, int col, float v) {
  int which  = col >> 12;        // /4096 : 0=q 1=k 2=v
  int within = col & 4095;
  int head   = within >> 7;      // /128
  int d      = within & 127;
  int b      = row >> 11;        // /2048
  int s      = row & 2047;
  size_t off = (size_t)which * PER_BUF +
               (size_t)((b << 5) + head) * (SS * HDD) + (size_t)s * HDD + d;
  qkv[off] = f2bf(v);
}

__global__ __launch_bounds__(256) void k_gemm_qkv(const __bf16* __restrict__ A,
                                                  const __bf16* __restrict__ Bm,
                                                  __bf16* __restrict__ qkv) {
  __shared__ alignas(16) __bf16 As[64 * 32];
  __shared__ alignas(16) __bf16 Bst[64 * 32];
  const int m_blk = blockIdx.y * 64, n_blk = blockIdx.x * 64;
  v8f acc0, acc1;
  gemm_mainloop(A, Bm, HH, N3, m_blk, n_blk, As, Bst, acc0, acc1);
  const int lane = threadIdx.x & 31, w = threadIdx.x >> 5;
  const int wm = w >> 1, wn = w & 1;
  const int r = lane & 15, hf = lane >> 4;
  const int mbase = m_blk + wm * 16 + hf * 8;
  const int col0  = n_blk + wn * 32 + r;
#pragma unroll
  for (int j = 0; j < 8; ++j) {
    qkv_scatter(qkv, mbase + j, col0,      acc0[j]);
    qkv_scatter(qkv, mbase + j, col0 + 16, acc1[j]);
  }
}

// ---------------- output projection GEMM (f32 out) ----------------
__global__ __launch_bounds__(256) void k_gemm_out(const __bf16* __restrict__ A,
                                                  const __bf16* __restrict__ Bm,
                                                  float* __restrict__ out) {
  __shared__ alignas(16) __bf16 As[64 * 32];
  __shared__ alignas(16) __bf16 Bst[64 * 32];
  const int m_blk = blockIdx.y * 64, n_blk = blockIdx.x * 64;
  v8f acc0, acc1;
  gemm_mainloop(A, Bm, HH, HH, m_blk, n_blk, As, Bst, acc0, acc1);
  const int lane = threadIdx.x & 31, w = threadIdx.x >> 5;
  const int wm = w >> 1, wn = w & 1;
  const int r = lane & 15, hf = lane >> 4;
  const int mbase = m_blk + wm * 16 + hf * 8;
  const int col0  = n_blk + wn * 32 + r;
#pragma unroll
  for (int j = 0; j < 8; ++j) {
    out[(size_t)(mbase + j) * HH + col0]      = acc0[j];
    out[(size_t)(mbase + j) * HH + col0 + 16] = acc1[j];
  }
}

// ---------------- RoPE (in place on bf16 Q and K) ----------------
__global__ void k_rope(const int* __restrict__ pos, __bf16* __restrict__ Qb,
                       __bf16* __restrict__ Kb) {
  int idx = blockIdx.x * blockDim.x + threadIdx.x;   // 2 * 64 * 2048 * 64 threads
  int which = idx >> 23;
  int rem   = idx & 8388607;
  int hl    = rem >> 17;          // (b*NH + head)
  int r2    = rem & 131071;
  int s     = r2 >> 6;
  int d     = r2 & 63;
  int b     = hl >> 5;
  float fp  = (float)pos[b * SS + s];
  // inv_freq = 10000^(-d/64) = exp2(-d * log2(10000)/64)
  float ang = fp * exp2f(-(float)d * 0.2076205059304601f);
  float c, sn;
  __sincosf(ang, &sn, &c);
  __bf16* base = (which ? Kb : Qb) + (size_t)hl * (SS * HDD) + (size_t)s * HDD;
  float x1 = bf2f(base[d]);
  float x2 = bf2f(base[d + 64]);
  base[d]      = f2bf(x1 * c - x2 * sn);
  base[d + 64] = f2bf(x2 * c + x1 * sn);
}

// ---------------- flash attention ----------------
// grid: (S/64 q-tiles, B*NH). 128 threads = 4 waves; wave w owns q rows
// [qt0 + 16w, +16). Online softmax, key tiles of 32. K tile via TDM.
__global__ __launch_bounds__(128) void k_flash(const __bf16* __restrict__ Qb,
                                               const __bf16* __restrict__ Kb,
                                               const __bf16* __restrict__ Vb,
                                               __bf16* __restrict__ attn) {
  __shared__ alignas(16) __bf16 Kt[32 * 128];   // [key][d]
  __shared__ alignas(16) __bf16 Vt[128 * 32];   // [d][key] (transposed)
  __shared__ alignas(16) __bf16 Pt[4 * 16 * 32];
  const float scale = 0.08838834764831845f;     // 128^-0.5
  const int tid = threadIdx.x, lane = tid & 31, w = tid >> 5;
  const int r = lane & 15, hf = lane >> 4;
  const int hl  = blockIdx.y;                   // b*NH + head
  const int qt0 = blockIdx.x * 64;
  const int q0  = qt0 + w * 16;
  const __bf16* Qh = Qb + (size_t)hl * (SS * HDD);
  const __bf16* Kh = Kb + (size_t)hl * (SS * HDD);
  const __bf16* Vh = Vb + (size_t)hl * (SS * HDD);

  // Q fragments held in registers: 4 chunks of 16x32 (d)
  v16bf qf[4];
#pragma unroll
  for (int c = 0; c < 4; ++c) {
    const __bf16* p = Qh + (size_t)(q0 + r) * HDD + c * 32 + hf * 16;
    qf[c] = cat8(*(const v8bf*)p, *(const v8bf*)(p + 8));
  }
  float Mx[8], Lx[8];
  v8f O[8];
#pragma unroll
  for (int j = 0; j < 8; ++j) { Mx[j] = -1e30f; Lx[j] = 0.0f; O[j] = v8f_zero(); }

  const int kend = qt0 + 64;                    // causal upper bound for block
  for (int k0 = 0; k0 < kend; k0 += 32) {
#if HAVE_TDM
    if (w == 0) {
      tdm_load_2d((unsigned)(size_t)Kt, Kh + (size_t)k0 * HDD,
                  /*tile*/HDD, 32, /*tensor*/HDD, 32, /*stride*/HDD);
    }
#endif
    // cooperative V load (transposed into LDS)
#pragma unroll
    for (int p = 0; p < 4; ++p) {
      int idx  = p * 128 + tid;
      int krow = idx >> 4;
      int cg   = (idx & 15) << 3;
#if !HAVE_TDM
      *(uint4*)(Kt + krow * HDD + cg) =
          *(const uint4*)(Kh + (size_t)(k0 + krow) * HDD + cg);
#endif
      union { uint4 u; __bf16 h[8]; } vu;
      vu.u = *(const uint4*)(Vh + (size_t)(k0 + krow) * HDD + cg);
#pragma unroll
      for (int i = 0; i < 8; ++i) Vt[(cg + i) * 32 + krow] = vu.h[i];
      if (k0 + 32 < kend)
        __builtin_prefetch(Vh + (size_t)(k0 + 32 + krow) * HDD + cg, 0, 1);
    }
#if HAVE_TDM
    if (w == 0) __builtin_amdgcn_s_wait_tensorcnt(0);
#endif
    __syncthreads();
    if (k0 <= q0 + 15) {                        // wave-uniform causal skip
      // S = Q K^T : two 16x16 subtiles, K-dim = HD in 4 chunks of 32
      v8f s0 = v8f_zero(), s1 = v8f_zero();
      const int kb = hf * 16;
#pragma unroll
      for (int c = 0; c < 4; ++c) {
        const __bf16* p0 = Kt + (size_t)r * HDD + c * 32 + kb;
        const __bf16* p1 = Kt + (size_t)(16 + r) * HDD + c * 32 + kb;
        s0 = wmma32(qf[c], cat8(*(const v8bf*)p0, *(const v8bf*)(p0 + 8)), s0);
        s1 = wmma32(qf[c], cat8(*(const v8bf*)p1, *(const v8bf*)(p1 + 8)), s1);
      }
      __bf16* Pw = Pt + w * (16 * 32);
#pragma unroll
      for (int j = 0; j < 8; ++j) {
        const int qrow = q0 + j + hf * 8;       // D layout: M = j + (lane/16)*8
        float x0 = s0[j] * scale; if (k0 + r > qrow)      x0 = -1e30f;
        float x1 = s1[j] * scale; if (k0 + 16 + r > qrow) x1 = -1e30f;
        float mt = fmaxf(x0, x1);
        mt = fmaxf(mt, __shfl_xor(mt, 1));
        mt = fmaxf(mt, __shfl_xor(mt, 2));
        mt = fmaxf(mt, __shfl_xor(mt, 4));
        mt = fmaxf(mt, __shfl_xor(mt, 8));
        float Mn   = fmaxf(Mx[j], mt);
        float corr = __expf(Mx[j] - Mn);
        float p0 = __expf(x0 - Mn);
        float p1 = __expf(x1 - Mn);
        float ls = p0 + p1;
        ls += __shfl_xor(ls, 1);
        ls += __shfl_xor(ls, 2);
        ls += __shfl_xor(ls, 4);
        ls += __shfl_xor(ls, 8);
        Lx[j] = Lx[j] * corr + ls;
        Mx[j] = Mn;
#pragma unroll
        for (int nd = 0; nd < 8; ++nd) O[nd][j] *= corr;
        // P to LDS (D layout) for re-read in A layout
        Pw[(j + hf * 8) * 32 + r]      = f2bf(p0);
        Pw[(j + hf * 8) * 32 + 16 + r] = f2bf(p1);
      }
      // P A-fragment (same-wave LDS ops are in-order)
      v16bf pa = cat8(*(const v8bf*)(Pw + r * 32 + hf * 16),
                      *(const v8bf*)(Pw + r * 32 + hf * 16 + 8));
#pragma unroll
      for (int nd = 0; nd < 8; ++nd) {
        const __bf16* pv = Vt + (nd * 16 + r) * 32 + kb;
        O[nd] = wmma32(pa, cat8(*(const v8bf*)pv, *(const v8bf*)(pv + 8)), O[nd]);
      }
    }
    __syncthreads();
  }
  // epilogue: attn[b, s, head*HD + d] bf16, row-major [B,S,H]
  const int b = hl >> 5, head = hl & 31;
#pragma unroll
  for (int j = 0; j < 8; ++j) {
    const int row = q0 + j + hf * 8;
    const float invl = 1.0f / Lx[j];
    __bf16* dst = attn + (size_t)(b * SS + row) * HH + head * HDD;
#pragma unroll
    for (int nd = 0; nd < 8; ++nd) dst[nd * 16 + r] = f2bf(O[nd][j] * invl);
  }
}

// ---------------- launch ----------------
extern "C" void kernel_launch(void* const* d_in, const int* in_sizes, int n_in,
                              void* d_out, int out_size, void* d_ws, size_t ws_size,
                              hipStream_t stream) {
  (void)in_sizes; (void)n_in; (void)out_size; (void)ws_size;
  const int*   positions = (const int*)d_in[0];
  const float* hidden    = (const float*)d_in[1];
  const float* wqkv      = (const float*)d_in[2];
  const float* wout      = (const float*)d_in[3];
  float* out = (float*)d_out;
  char*  ws  = (char*)d_ws;

  // workspace layout (bytes)
  __bf16* wqkv_bf = (__bf16*)(ws);                 // 100,663,296 B
  __bf16* wout_bf = (__bf16*)(ws + 100663296);     //  33,554,432 B
  __bf16* hid_bf  = (__bf16*)(ws + 134217728);     //  33,554,432 B
  __bf16* qkv_bf  = (__bf16*)(ws + 167772160);     // 3 x 33,554,432 B (Q|K|V head-major)
  __bf16* attn_bf = (__bf16*)(ws + 268435456);     //  33,554,432 B

  k_cvt<<<(HH * N3 + 255) / 256, 256, 0, stream>>>(wqkv, wqkv_bf, HH * N3);
  k_cvt<<<(HH * HH + 255) / 256, 256, 0, stream>>>(wout, wout_bf, HH * HH);
  k_cvt<<<(MM * HH + 255) / 256, 256, 0, stream>>>(hidden, hid_bf, MM * HH);

  k_gemm_qkv<<<dim3(N3 / 64, MM / 64), 256, 0, stream>>>(hid_bf, wqkv_bf, qkv_bf);

  k_rope<<<(2 * HEADS * SS * 64) / 256, 256, 0, stream>>>(positions, qkv_bf,
                                                          qkv_bf + PER_BUF);

  k_flash<<<dim3(SS / 64, HEADS), 128, 0, stream>>>(qkv_bf, qkv_bf + PER_BUF,
                                                    qkv_bf + 2 * PER_BUF, attn_bf);

  k_gemm_out<<<dim3(HH / 64, MM / 64), 256, 0, stream>>>(attn_bf, wout_bf, out);
}
